// CGCNNPyGChargeEarlyCorrected_74637941670359
// MI455X (gfx1250) — compile-verified, matrix operand order checked by reading, not run on hardware
//
#include <hip/hip_runtime.h>
#include <hip/hip_bf16.h>

// ---------------------------------------------------------------------------
// CGCNN forward for MI455X (gfx1250, wave32, WMMA + async-to-LDS).
//
//   1) node embed   : x = [x92 | chargeEmb(batch)] @ Wa + ba     (f32 + f16 shadow)
//   2) bond embed   : ea = edge_attr @ Wb + bb                   (f16)
//   3) 3x conv layer: fused 4-GEMM edge kernel, A tiles staged with
//      global_load_async_to_lds_b128, GEMMs via v_wmma_f32_16x16x32_f16,
//      f32 atomic scatter into x_new; then BN+softplus+residual (updates shadow)
//   4) mean pool + tiny predictor MLP
// ---------------------------------------------------------------------------

typedef __attribute__((ext_vector_type(16))) _Float16 v16h;
typedef __attribute__((ext_vector_type(8)))  _Float16 v8h;
typedef __attribute__((ext_vector_type(8)))  float    v8f;

#define LDS_FENCE()   asm volatile("s_wait_dscnt 0" ::: "memory")
#define ASYNC_FENCE() asm volatile("s_wait_asynccnt 0" ::: "memory")

__device__ __forceinline__ float sp_f(float v) {
    // numerically stable softplus: max(v,0) + log1p(exp(-|v|))
    float a = fabsf(v);
    return fmaxf(v, 0.f) + log1pf(__expf(-a));
}

// --- WMMA fragment loaders --------------------------------------------------
// A fragment (16x32 f16): lane<16 holds row=lane, K = kb+0..7 and kb+16..23;
// lane>=16 holds row=lane-16, K = kb+8..15 and kb+24..31.
__device__ __forceinline__ v16h load_afrag(const _Float16* aT, int lda, int kb, int lane) {
    const _Float16* p = aT + (lane & 15) * lda + kb + ((lane >> 4) << 3);
    v8h lo = *(const v8h*)(p);
    v8h hi = *(const v8h*)(p + 16);
    return __builtin_shufflevector(lo, hi, 0,1,2,3,4,5,6,7,8,9,10,11,12,13,14,15);
}
// B fragment: pre-swizzled, 16 contiguous halves per lane.
__device__ __forceinline__ v16h load_bfrag(const _Float16* p) {
    v8h lo = *(const v8h*)(p);
    v8h hi = *(const v8h*)(p + 8);
    return __builtin_shufflevector(lo, hi, 0,1,2,3,4,5,6,7,8,9,10,11,12,13,14,15);
}

template <int KS>
__device__ __forceinline__ void mma_block(const _Float16* aT, int lda,
                                          const _Float16* wf, int fragBase,
                                          int lane, v8f acc[4]) {
#pragma unroll
    for (int ks = 0; ks < KS; ++ks) {
        v16h a = load_afrag(aT, lda, ks * 32, lane);
#pragma unroll
        for (int nt = 0; nt < 4; ++nt) {
            v16h b = load_bfrag(wf + (size_t)(((fragBase + ks * 4 + nt) << 5) | lane) * 16);
            acc[nt] = __builtin_amdgcn_wmma_f32_16x16x32_f16(
                false, a, false, b, (short)0, acc[nt], false, false);
        }
    }
}

// ---------------------------------------------------------------------------
// Weight pre-swizzle: f32 weights -> f16 WMMA B fragments in global scratch.
// Frag space per layer: eW1 frags 0..23 (6 ks x 4 nt), eW2 24..31,
// nW1 32..47, nW2 48..55. Each frag = 32 lanes x 16 halves (contiguous).
// ---------------------------------------------------------------------------
__global__ void prep_wfrag_kernel(const float* __restrict__ eW1,
                                  const float* __restrict__ eW2,
                                  const float* __restrict__ nW1,
                                  const float* __restrict__ nW2,
                                  _Float16* __restrict__ wfrag) {
    int idx = blockIdx.x * blockDim.x + threadIdx.x;
    const int total = 3 * 56 * 512;
    if (idx >= total) return;
    int layer = idx / (56 * 512);
    int rem   = idx % (56 * 512);
    int frag  = rem / 512;
    int lane  = (rem % 512) / 16;
    int h     = rem % 16;

    const float* W;
    int fb;
    if (frag < 24)      { W = eW1 + layer * 192 * 64; fb = 0;  }
    else if (frag < 32) { W = eW2 + layer * 64 * 64;  fb = 24; }
    else if (frag < 48) { W = nW1 + layer * 128 * 64; fb = 32; }
    else                { W = nW2 + layer * 64 * 64;  fb = 48; }
    int lf = frag - fb;
    int ks = lf >> 2, nt = lf & 3;
    int klocal = (h < 8) ? h : (h + 8);                 // 0..7 or 16..23
    int k = ks * 32 + klocal + ((lane >= 16) ? 8 : 0);  // mirrors A layout
    int col = nt * 16 + (lane & 15);
    wfrag[idx] = (_Float16)W[k * 64 + col];
}

// ---------------------------------------------------------------------------
// Embeddings (node embed also writes the f16 shadow used by the edge kernel)
// ---------------------------------------------------------------------------
__global__ void node_embed_kernel(const float* __restrict__ xin,
                                  const float* __restrict__ charge,
                                  const int* __restrict__ batch,
                                  const float* __restrict__ Wc,
                                  const float* __restrict__ bc,
                                  const float* __restrict__ Wa,
                                  const float* __restrict__ ba,
                                  float* __restrict__ xout,
                                  _Float16* __restrict__ xh, int nNodes) {
    int i = blockIdx.x * blockDim.x + threadIdx.x;
    if (i >= nNodes * 64) return;
    int n = i >> 6, f = i & 63;
    float s = ba[f];
    const float* xr = xin + (size_t)n * 92;
    for (int k = 0; k < 92; ++k) s += xr[k] * Wa[k * 64 + f];
    float ch = charge[batch[n]];
    for (int c = 0; c < 16; ++c) s += (ch * Wc[c] + bc[c]) * Wa[(92 + c) * 64 + f];
    xout[i] = s;
    xh[i] = (_Float16)s;
}

__global__ void bond_embed_kernel(const float* __restrict__ eattr,
                                  const float* __restrict__ Wb,
                                  const float* __restrict__ bb,
                                  _Float16* __restrict__ ea, int nE) {
    int i = blockIdx.x * blockDim.x + threadIdx.x;
    if (i >= nE * 64) return;
    int e = i >> 6, f = i & 63;
    float s = bb[f];
    const float* ar = eattr + (size_t)e * 41;
    for (int k = 0; k < 41; ++k) s += ar[k] * Wb[k * 64 + f];
    ea[i] = (_Float16)s;
}

// ---------------------------------------------------------------------------
// Fused per-edge conv layer: 4 GEMMs via WMMA + atomic scatter.
// One wave per 16-edge tile; 4 waves per block; grid-stride over tiles.
// A tile [xr | xc | ea] is f16 in LDS, staged with async-to-LDS copies.
// ---------------------------------------------------------------------------
#define CONV_WAVES 4

__global__ __launch_bounds__(128) void edge_conv_kernel(
    const _Float16* __restrict__ xh, _Float16* __restrict__ ea,
    float* __restrict__ xnew,
    const int* __restrict__ rowI, const int* __restrict__ colI,
    const _Float16* __restrict__ wfrag,
    const float* __restrict__ eb1, const float* __restrict__ eb2,
    const float* __restrict__ nb1, const float* __restrict__ nb2,
    int nEdges) {
    __shared__ __align__(16) _Float16 sA[CONV_WAVES][16 * 192]; // [xr|xc|ea] tile
    __shared__ __align__(16) _Float16 sH[CONV_WAVES][16 * 64];  // hidden tile
    __shared__ float sBias[256];
    __shared__ int sRow[CONV_WAVES][16];
    __shared__ int sCol[CONV_WAVES][16];

    const int tid = threadIdx.x;
    if (tid < 64) {
        sBias[tid]       = eb1[tid];
        sBias[64 + tid]  = eb2[tid];
        sBias[128 + tid] = nb1[tid];
        sBias[192 + tid] = nb2[tid];
    }
    __syncthreads();

    const int lane = tid & 31;
    const int wv   = tid >> 5;
    _Float16* aT = sA[wv];
    _Float16* hT = sH[wv];
    // LDS byte address of this wave's A tile (generic LDS pointers carry the
    // DS byte offset in their low 32 bits on gfx1250's flat aperture scheme)
    const unsigned aBase = (unsigned)(unsigned long long)(void*)aT;

    const int nTiles = (nEdges + 15) >> 4;
    const int gwave  = blockIdx.x * CONV_WAVES + wv;
    const int nwaves = gridDim.x * CONV_WAVES;

    const int cb = lane & 15;          // column within 16-wide N tile
    const int rb = (lane >> 4) << 3;   // row base (0 or 8) of C fragment

    for (int tile = gwave; tile < nTiles; tile += nwaves) {
        const int e0 = tile << 4;

        // prefetch next tile's index + edge-feature lines while we work
        int nextTile = tile + nwaves;
        if (nextTile < nTiles) {
            __builtin_prefetch(rowI + (nextTile << 4), 0, 3);
            __builtin_prefetch(colI + (nextTile << 4), 0, 3);
            __builtin_prefetch(ea + ((size_t)nextTile << 10), 0, 3);
        }

        // --- stage edge endpoint indices ---
        if (lane < 16) {
            int e = e0 + lane;
            bool ok = e < nEdges;
            sRow[wv][lane] = ok ? rowI[e] : -1;
            sCol[wv][lane] = ok ? colI[e] : -1;
        }
        LDS_FENCE();

        if (e0 + 16 <= nEdges) {
            // --- async byte-copy staging: xr | xc | ea (each 16 rows x 128B) ---
            // 128 x 16B chunks per sub-tile; 4 rounds of 32 lanes each.
#pragma unroll
            for (int j = 0; j < 4; ++j) {
                int chunk = (j << 5) + lane;        // 0..127
                int rr    = chunk >> 3;             // row (8 chunks per 128B row)
                int cby   = (chunk & 7) << 4;       // byte offset within row
                int nr = sRow[wv][rr];
                int nc = sCol[wv][rr];
                unsigned l0 = aBase + rr * 384 + cby;        // xr slot
                unsigned l1 = l0 + 128;                      // xc slot
                unsigned l2 = l0 + 256;                      // ea slot
                const _Float16* g0 = xh + ((size_t)nr << 6) + ((chunk & 7) << 3);
                const _Float16* g1 = xh + ((size_t)nc << 6) + ((chunk & 7) << 3);
                const _Float16* g2 = ea + ((size_t)e0 << 6) + ((size_t)chunk << 3);
                asm volatile("global_load_async_to_lds_b128 %0, %1, off"
                             :: "v"(l0), "v"(g0) : "memory");
                asm volatile("global_load_async_to_lds_b128 %0, %1, off"
                             :: "v"(l1), "v"(g1) : "memory");
                asm volatile("global_load_async_to_lds_b128 %0, %1, off"
                             :: "v"(l2), "v"(g2) : "memory");
            }
            ASYNC_FENCE();
        } else {
            // --- tail-tile fallback: per-element staging with zero padding ---
            for (int i = lane; i < 16 * 64; i += 32) {
                int rr = i >> 6, cc = i & 63;
                int nr = sRow[wv][rr];
                int nc = sCol[wv][rr];
                _Float16 vr = (nr >= 0) ? xh[((size_t)nr << 6) + cc] : (_Float16)0.f;
                _Float16 vc = (nc >= 0) ? xh[((size_t)nc << 6) + cc] : (_Float16)0.f;
                _Float16 ve = (nr >= 0) ? ea[((size_t)(e0 + rr) << 6) + cc] : (_Float16)0.f;
                aT[rr * 192 + cc]       = vr;
                aT[rr * 192 + 64 + cc]  = vc;
                aT[rr * 192 + 128 + cc] = ve;
            }
        }
        LDS_FENCE();

        v8f acc[4];
        const v8f vzero = {};

        // ===== MLP1: h1 = softplus([xr|xc|ea] @ eW1 + eb1)   (K=192) =====
        acc[0] = vzero; acc[1] = vzero; acc[2] = vzero; acc[3] = vzero;
        mma_block<6>(aT, 192, wfrag, 0, lane, acc);
#pragma unroll
        for (int nt = 0; nt < 4; ++nt) {
            int col = nt * 16 + cb;
            float bi = sBias[col];
#pragma unroll
            for (int r = 0; r < 8; ++r)
                hT[(rb + r) * 64 + col] = (_Float16)sp_f(acc[nt][r] + bi);
        }
        LDS_FENCE();

        // ===== MLP2: ea_new = h1 @ eW2 + eb2                 (K=64) =====
        acc[0] = vzero; acc[1] = vzero; acc[2] = vzero; acc[3] = vzero;
        mma_block<2>(hT, 64, wfrag, 24, lane, acc);
#pragma unroll
        for (int nt = 0; nt < 4; ++nt) {
            int col = nt * 16 + cb;
            float bi = sBias[64 + col];
#pragma unroll
            for (int r = 0; r < 8; ++r) {
                int rrow = rb + r;
                float v = acc[nt][r] + bi;
                aT[rrow * 192 + 64 + col] = (_Float16)v;   // overwrite xc slot
                int e = e0 + rrow;
                if (e < nEdges) ea[((size_t)e << 6) + col] = (_Float16)v;
            }
        }
        LDS_FENCE();

        // ===== MLP3: h2 = softplus([xr|ea_new] @ nW1 + nb1)  (K=128) =====
        acc[0] = vzero; acc[1] = vzero; acc[2] = vzero; acc[3] = vzero;
        mma_block<4>(aT, 192, wfrag, 32, lane, acc);
#pragma unroll
        for (int nt = 0; nt < 4; ++nt) {
            int col = nt * 16 + cb;
            float bi = sBias[128 + col];
#pragma unroll
            for (int r = 0; r < 8; ++r)
                hT[(rb + r) * 64 + col] = (_Float16)sp_f(acc[nt][r] + bi);
        }
        LDS_FENCE();

        // ===== MLP4: msg = h2 @ nW2 + nb2, scatter to x_new  (K=64) =====
        acc[0] = vzero; acc[1] = vzero; acc[2] = vzero; acc[3] = vzero;
        mma_block<2>(hT, 64, wfrag, 48, lane, acc);
#pragma unroll
        for (int nt = 0; nt < 4; ++nt) {
            int col = nt * 16 + cb;
            float bi = sBias[192 + col];
#pragma unroll
            for (int r = 0; r < 8; ++r) {
                int rrow = rb + r;
                int cn = sCol[wv][rrow];
                if (cn >= 0)
                    atomicAdd(&xnew[((size_t)cn << 6) + col], acc[nt][r] + bi);
            }
        }
        LDS_FENCE();   // all DS reads done before next tile's async writes land
    }
}

// ---------------------------------------------------------------------------
// BatchNorm stats (sum, sumsq per feature) + apply(+softplus+residual)
// ---------------------------------------------------------------------------
__global__ void bn_stats_kernel(const float* __restrict__ xn,
                                float* __restrict__ stats, int nNodes) {
    __shared__ float rs[256], rq[256];
    int t = threadIdx.x;
    int f = t & 63, rg = t >> 6;
    float s = 0.f, q = 0.f;
    for (int n = blockIdx.x * 4 + rg; n < nNodes; n += gridDim.x * 4) {
        float v = xn[(size_t)n * 64 + f];
        s += v; q += v * v;
    }
    rs[t] = s; rq[t] = q;
    __syncthreads();
    if (t < 64) {
        s = rs[t] + rs[64 + t] + rs[128 + t] + rs[192 + t];
        q = rq[t] + rq[64 + t] + rq[128 + t] + rq[192 + t];
        atomicAdd(&stats[f], s);
        atomicAdd(&stats[64 + f], q);
    }
}

__global__ void bn_apply_kernel(float* __restrict__ x, _Float16* __restrict__ xh,
                                const float* __restrict__ xn,
                                const float* __restrict__ stats,
                                const float* __restrict__ gamma,
                                const float* __restrict__ beta,
                                int total, float invN) {
    int i = blockIdx.x * blockDim.x + threadIdx.x;
    if (i >= total) return;
    int f = i & 63;
    float mean = stats[f] * invN;
    float var  = stats[64 + f] * invN - mean * mean;
    float v = (xn[i] - mean) * rsqrtf(var + 1e-5f) * gamma[f] + beta[f];
    float r = sp_f(v) + x[i];
    x[i] = r;
    xh[i] = (_Float16)r;   // keep f16 shadow in sync for next layer's gathers
}

// ---------------------------------------------------------------------------
// Pooling + predictor
// ---------------------------------------------------------------------------
__global__ void pool_kernel(const float* __restrict__ x, const int* __restrict__ batch,
                            float* __restrict__ sums, float* __restrict__ cnt, int total) {
    int i = blockIdx.x * blockDim.x + threadIdx.x;
    if (i >= total) return;
    int n = i >> 6, f = i & 63;
    int b = batch[n];
    atomicAdd(&sums[(size_t)b * 64 + f], x[i]);
    if (f == 0) atomicAdd(&cnt[b], 1.0f);
}

__global__ __launch_bounds__(128) void predict_kernel(
    const float* __restrict__ sums, const float* __restrict__ cnt,
    const float* __restrict__ pW1, const float* __restrict__ pb1,
    const float* __restrict__ pW2, const float* __restrict__ pb2,
    const float* __restrict__ pW3, const float* __restrict__ pb3,
    float* __restrict__ out) {
    __shared__ float g[64];
    __shared__ float h1[128];
    __shared__ float h2[128];
    __shared__ float red[128];
    int b = blockIdx.x, t = threadIdx.x;
    if (t < 64) g[t] = sums[(size_t)b * 64 + t] / fmaxf(cnt[b], 1.0f);
    __syncthreads();
    float s = pb1[t];
    for (int k = 0; k < 64; ++k) s += g[k] * pW1[k * 128 + t];
    h1[t] = sp_f(s);
    __syncthreads();
    s = pb2[t];
    for (int k = 0; k < 128; ++k) s += h1[k] * pW2[k * 128 + t];
    h2[t] = sp_f(s);
    __syncthreads();
    red[t] = h2[t] * pW3[t];
    __syncthreads();
    for (int off = 64; off > 0; off >>= 1) {
        if (t < off) red[t] += red[t + off];
        __syncthreads();
    }
    if (t == 0) out[b] = red[0] + pb3[0];
}

__global__ void fill_kernel(float* __restrict__ p, int n) {
    int i = blockIdx.x * blockDim.x + threadIdx.x;
    if (i < n) p[i] = 0.f;
}

// ---------------------------------------------------------------------------
// Launch
// ---------------------------------------------------------------------------
extern "C" void kernel_launch(void* const* d_in, const int* in_sizes, int n_in,
                              void* d_out, int out_size, void* d_ws, size_t ws_size,
                              hipStream_t stream) {
    (void)n_in; (void)out_size; (void)ws_size;
    const float* x_in   = (const float*)d_in[0];
    const float* eattr  = (const float*)d_in[1];
    const float* charge = (const float*)d_in[2];
    const int*   eidx   = (const int*)d_in[3];
    const int*   batch  = (const int*)d_in[4];
    const float* Wc  = (const float*)d_in[5];
    const float* bc  = (const float*)d_in[6];
    const float* Wa  = (const float*)d_in[7];
    const float* ba  = (const float*)d_in[8];
    const float* Wb  = (const float*)d_in[9];
    const float* bb  = (const float*)d_in[10];
    const float* eW1 = (const float*)d_in[11];
    const float* eb1 = (const float*)d_in[12];
    const float* eW2 = (const float*)d_in[13];
    const float* eb2 = (const float*)d_in[14];
    const float* nW1 = (const float*)d_in[15];
    const float* nb1 = (const float*)d_in[16];
    const float* nW2 = (const float*)d_in[17];
    const float* nb2 = (const float*)d_in[18];
    const float* gam = (const float*)d_in[19];
    const float* bet = (const float*)d_in[20];
    const float* pW1 = (const float*)d_in[21];
    const float* pb1 = (const float*)d_in[22];
    const float* pW2 = (const float*)d_in[23];
    const float* pb2 = (const float*)d_in[24];
    const float* pW3 = (const float*)d_in[25];
    const float* pb3 = (const float*)d_in[26];

    const int nN = in_sizes[0] / 92;
    const int nE = in_sizes[1] / 41;
    const int nG = in_sizes[2];

    // workspace carve-out
    char* ws = (char*)d_ws;
    size_t off = 0;
    auto take = [&](size_t bytes) -> void* {
        void* p = ws + off;
        off += (bytes + 255) & ~(size_t)255;
        return p;
    };
    _Float16* ea    = (_Float16*)take((size_t)nE * 64 * 2);
    float*    xf    = (float*)take((size_t)nN * 64 * 4);
    _Float16* xh    = (_Float16*)take((size_t)nN * 64 * 2);
    float*    xnew  = (float*)take((size_t)nN * 64 * 4);
    _Float16* wfrag = (_Float16*)take((size_t)3 * 56 * 512 * 2);
    float*    stats = (float*)take(128 * 4);
    float*    psum  = (float*)take((size_t)nG * 64 * 4);
    float*    pcnt  = (float*)take((size_t)nG * 4);

    const int* rowI = eidx;
    const int* colI = eidx + nE;

    // 0) weight pre-swizzle into WMMA B-fragment layout (f16)
    {
        int total = 3 * 56 * 512;
        prep_wfrag_kernel<<<(total + 255) / 256, 256, 0, stream>>>(eW1, eW2, nW1, nW2, wfrag);
    }
    // 1) embeddings
    node_embed_kernel<<<(nN * 64 + 255) / 256, 256, 0, stream>>>(
        x_in, charge, batch, Wc, bc, Wa, ba, xf, xh, nN);
    bond_embed_kernel<<<(nE * 64 + 255) / 256, 256, 0, stream>>>(eattr, Wb, bb, ea, nE);

    // 2) conv layers
    for (int L = 0; L < 3; ++L) {
        fill_kernel<<<(nN * 64 + 255) / 256, 256, 0, stream>>>(xnew, nN * 64);
        fill_kernel<<<1, 128, 0, stream>>>(stats, 128);
        edge_conv_kernel<<<2048, 128, 0, stream>>>(
            xh, ea, xnew, rowI, colI,
            wfrag + (size_t)L * 56 * 512,
            eb1 + L * 64, eb2 + L * 64, nb1 + L * 64, nb2 + L * 64, nE);
        bn_stats_kernel<<<256, 256, 0, stream>>>(xnew, stats, nN);
        bn_apply_kernel<<<(nN * 64 + 255) / 256, 256, 0, stream>>>(
            xf, xh, xnew, stats, gam + L * 64, bet + L * 64, nN * 64, 1.0f / (float)nN);
    }

    // 3) pooling + predictor
    fill_kernel<<<(nG * 64 + 255) / 256, 256, 0, stream>>>(psum, nG * 64);
    fill_kernel<<<1, 128, 0, stream>>>(pcnt, nG);
    pool_kernel<<<(nN * 64 + 255) / 256, 256, 0, stream>>>(xf, batch, psum, pcnt, nN * 64);
    predict_kernel<<<nG, 128, 0, stream>>>(psum, pcnt, pW1, pb1, pW2, pb2, pW3, pb3,
                                           (float*)d_out);
}
